// Tacotron2_74148315398761
// MI455X (gfx1250) — matrix operand order repaired and checked by reading, not compiled, not run on hardware
//
#include <hip/hip_runtime.h>
#include <hip/hip_bf16.h>
#include <math.h>

// ---------------- Problem constants (from setup_inputs) ----------------
#define B_    64
#define S_    256
#define T_    500
#define NMEL  80
#define E_    512
#define P_    256
#define A_    1024
#define D_    1024
#define AD_   128
#define F_    32
#define KW_   31
#define NMELP 96   // 80 padded to multiple of 32

typedef __attribute__((ext_vector_type(16))) __bf16 v16bf;
typedef __attribute__((ext_vector_type(8)))  float  v8f;

union FragBF {
  v16bf m;
  uint4 q[2];
};

static __device__ __forceinline__ unsigned short f2bf(float f) {
  unsigned int u = __builtin_bit_cast(unsigned int, f);
  u = (u + 0x7FFFu + ((u >> 16) & 1u)) >> 16;  // round-to-nearest-even
  return (unsigned short)u;
}

// ---------------- Generic multi-part WMMA bf16 GEMM ----------------
// Computes D[M,N] = sum_p A_p[M,K_p] * W_p^T  (+ bias) with optional relu,
// writing fp32 and/or bf16 outputs. W_p stored transposed: Wt[n*ldw + k].
// Block: 256 threads = 8 waves. Each wave owns a 32x16 output strip
// (two 16x16 accumulators sharing one B fragment -> 3 b128 loads per WMMA,
// 2x B reuse in registers). WG covers 64x64; grid = (N/64, M/64).
struct GemmArgs {
  const unsigned short* A[3];
  const unsigned short* W[3];
  int lda[3];
  int ldw[3];
  int K[3];
  const float* bias;   // [N] or null
  float* outF;         // [M,N] or null
  unsigned short* outH;// [M,N] bf16 or null
  int N;
  int relu;
};

__global__ __launch_bounds__(256) void k_wmma_gemm(GemmArgs ga) {
  const int lane = threadIdx.x & 31;
  const int wave = threadIdx.x >> 5;
  const int mh = (wave & 1) * 32;   // which 32-row half of the 64-row block
  const int nt = wave >> 1;         // 4 N strips per WG
  const int m0 = blockIdx.y * 64 + mh;
  const int n0 = blockIdx.x * 64 + nt * 16;
  const int lm = lane & 15;         // row/col within tile
  const int lg = lane >> 4;         // lane group 0/1

  v8f c0 = {}, c1 = {};

  for (int p = 0; p < 3; ++p) {
    const int K = ga.K[p];
    if (K == 0) continue;
    const int lda = ga.lda[p];
    const unsigned short* __restrict__ arow0 = ga.A[p] + (size_t)(m0 + lm) * lda;
    const unsigned short* __restrict__ arow1 = arow0 + (size_t)16 * lda;
    const unsigned short* __restrict__ brow  = ga.W[p] + (size_t)(n0 + lm) * ga.ldw[p];
    for (int k0 = 0; k0 < K; k0 += 32) {
      FragBF a0, a1, b;
      // B 32x16 bf16 layout (matches A): group0 lanes hold K k0..k0+7 and k0+16..k0+23,
      // group1 lanes hold k0+8..15 and k0+24..31.
      b.q[0] = *(const uint4*)(brow + k0 + lg * 8);
      b.q[1] = *(const uint4*)(brow + k0 + lg * 8 + 16);
      // A 16x32 bf16 layout: group0 lanes hold K k0..k0+7 and k0+16..k0+23,
      // group1 lanes hold k0+8..15 and k0+24..31 (elements 0..7 / 8..15).
      a0.q[0] = *(const uint4*)(arow0 + k0 + lg * 8);
      a0.q[1] = *(const uint4*)(arow0 + k0 + lg * 8 + 16);
      a1.q[0] = *(const uint4*)(arow1 + k0 + lg * 8);
      a1.q[1] = *(const uint4*)(arow1 + k0 + lg * 8 + 16);
      c0 = __builtin_amdgcn_wmma_f32_16x16x32_bf16(
          false, a0.m, false, b.m, (short)0, c0, false, false);
      c1 = __builtin_amdgcn_wmma_f32_16x16x32_bf16(
          false, a1.m, false, b.m, (short)0, c1, false, false);
    }
  }

  const int col = n0 + lm;
  const float bias = ga.bias ? ga.bias[col] : 0.f;
  #pragma unroll
  for (int r = 0; r < 8; ++r) {
    const int row = m0 + (lg ? 8 + r : r);
    float v0 = c0[r] + bias;
    float v1 = c1[r] + bias;
    if (ga.relu) { v0 = fmaxf(v0, 0.f); v1 = fmaxf(v1, 0.f); }
    const size_t o0 = (size_t)row * ga.N + col;
    const size_t o1 = (size_t)(row + 16) * ga.N + col;
    if (ga.outF) { ga.outF[o0] = v0; ga.outF[o1] = v1; }
    if (ga.outH) { ga.outH[o0] = f2bf(v0); ga.outH[o1] = f2bf(v1); }
  }
}

// ---------------- Conversion kernels ----------------
__global__ void k_transpose_bf(const float* __restrict__ W, unsigned short* __restrict__ Wt,
                               int K, int N, int Kpad) {
  int idx = blockIdx.x * 256 + threadIdx.x;
  if (idx >= N * Kpad) return;
  int n = idx / Kpad, k = idx - n * Kpad;
  Wt[idx] = (k < K) ? f2bf(W[(size_t)k * N + n]) : (unsigned short)0;
}

__global__ void k_convert_bf(const float* __restrict__ X, unsigned short* __restrict__ Y, int n) {
  int idx = blockIdx.x * 256 + threadIdx.x;
  if (idx < n) Y[idx] = f2bf(X[idx]);
}

__global__ void k_mel_slice(const float* __restrict__ mel, unsigned short* __restrict__ out, int t) {
  int idx = blockIdx.x * 256 + threadIdx.x;          // B_*NMELP
  if (idx >= B_ * NMELP) return;
  int b = idx / NMELP, k = idx - b * NMELP;
  out[idx] = (k < NMEL) ? f2bf(mel[((size_t)b * T_ + t) * NMEL + k]) : (unsigned short)0;
}

// ---------------- LSTM pointwise (gates i,f,g,o) ----------------
__global__ void k_lstm_pw(const float* __restrict__ z, float* __restrict__ c,
                          float* __restrict__ h, unsigned short* __restrict__ hbf, int H) {
  int idx = blockIdx.x * 256 + threadIdx.x;          // B_*H
  if (idx >= B_ * H) return;
  int b = idx / H, j = idx - b * H;
  const float* zb = z + (size_t)b * 4 * H;
  float gi = zb[j], gf = zb[H + j], gg = zb[2 * H + j], go = zb[3 * H + j];
  float si = 1.f / (1.f + expf(-gi));
  float sf = 1.f / (1.f + expf(-gf));
  float so = 1.f / (1.f + expf(-go));
  float c2 = sf * c[idx] + si * tanhf(gg);
  float h2 = so * tanhf(c2);
  c[idx] = c2; h[idx] = h2; hbf[idx] = f2bf(h2);
}

// ---------------- Location conv (K=31, SAME, 2->32 channels) ----------------
__global__ void k_locconv(const float* __restrict__ w, const float* __restrict__ wcum,
                          const float* __restrict__ cv, float* __restrict__ fbuf) {
  int idx = blockIdx.x * 256 + threadIdx.x;          // B_*S_*F_
  if (idx >= B_ * S_ * F_) return;
  int f = idx & (F_ - 1);
  int s = (idx >> 5) & (S_ - 1);
  int b = idx >> 13;
  float acc = 0.f;
  #pragma unroll
  for (int k = 0; k < KW_; ++k) {
    int sp = s + k - (KW_ / 2);
    if ((unsigned)sp < (unsigned)S_) {
      float a0 = w[b * S_ + sp], a1 = wcum[b * S_ + sp];
      acc += a0 * cv[k * (2 * F_) + f] + a1 * cv[k * (2 * F_) + F_ + f];
    }
  }
  fbuf[idx] = acc;
}

// ---------------- Attention energies ----------------
__global__ __launch_bounds__(AD_) void k_energy(const float* __restrict__ q,
                                                const float* __restrict__ keys,
                                                const float* __restrict__ fbuf,
                                                const float* __restrict__ wloc,
                                                const float* __restrict__ v,
                                                float* __restrict__ e) {
  int bs = blockIdx.x;                               // B_*S_
  int d = threadIdx.x;                               // AD_
  int b = bs >> 8;
  const float* fr = fbuf + (size_t)bs * F_;
  float fp = 0.f;
  #pragma unroll
  for (int f = 0; f < F_; ++f) fp += fr[f] * wloc[f * AD_ + d];
  float t = tanhf(q[b * AD_ + d] + keys[(size_t)bs * AD_ + d] + fp);
  float val = t * v[d];
  __shared__ float red[AD_];
  red[d] = val; __syncthreads();
  for (int off = AD_ / 2; off; off >>= 1) {
    if (d < off) red[d] += red[d + off];
    __syncthreads();
  }
  if (d == 0) e[bs] = red[0];
}

// ---------------- Softmax + cumulative + context (fused, per batch) ----------------
__global__ __launch_bounds__(S_) void k_softmax_ctx(const float* __restrict__ e,
                                                    float* __restrict__ w,
                                                    float* __restrict__ wcum,
                                                    float* __restrict__ aligns,
                                                    const float* __restrict__ memory,
                                                    float* __restrict__ ctx,
                                                    unsigned short* __restrict__ ctxbf,
                                                    int t) {
  int b = blockIdx.x, s = threadIdx.x;
  __shared__ float red[S_];
  __shared__ float sw[S_];
  float ev = e[b * S_ + s];
  red[s] = ev; __syncthreads();
  for (int off = S_ / 2; off; off >>= 1) {
    if (s < off) red[s] = fmaxf(red[s], red[s + off]);
    __syncthreads();
  }
  float mx = red[0]; __syncthreads();
  float ex = expf(ev - mx);
  red[s] = ex; __syncthreads();
  for (int off = S_ / 2; off; off >>= 1) {
    if (s < off) red[s] += red[s + off];
    __syncthreads();
  }
  float wn = ex / red[0];
  sw[s] = wn;
  w[b * S_ + s] = wn;
  wcum[b * S_ + s] += wn;
  aligns[((size_t)b * T_ + t) * S_ + s] = wn;
  __syncthreads();
  const float* mb = memory + (size_t)b * S_ * E_;
  for (int ee = s; ee < E_; ee += S_) {
    float acc = 0.f;
    for (int ss = 0; ss < S_; ++ss) acc += sw[ss] * mb[(size_t)ss * E_ + ee];
    ctx[b * E_ + ee] = acc;
    ctxbf[b * E_ + ee] = f2bf(acc);
  }
}

// ---------------- Projection + stop gate ----------------
__global__ __launch_bounds__(96) void k_proj(const float* __restrict__ h_d,
                                             const float* __restrict__ ctx,
                                             const float* __restrict__ pw,
                                             const float* __restrict__ pb,
                                             const float* __restrict__ gw,
                                             const float* __restrict__ gb,
                                             float* __restrict__ mels,
                                             float* __restrict__ stops, int t) {
  int b = blockIdx.x, tid = threadIdx.x;
  const float* hb = h_d + (size_t)b * D_;
  const float* cb = ctx + (size_t)b * E_;
  if (tid < NMEL) {
    float acc = pb[tid];
    for (int j = 0; j < D_; ++j)  acc += hb[j] * pw[(size_t)j * NMEL + tid];
    for (int j = 0; j < E_; ++j)  acc += cb[j] * pw[(size_t)(D_ + j) * NMEL + tid];
    mels[((size_t)b * T_ + t) * NMEL + tid] = acc;
  } else if (tid == NMEL) {
    float acc = gb[0];
    for (int j = 0; j < D_; ++j)  acc += hb[j] * gw[j];
    for (int j = 0; j < E_; ++j)  acc += cb[j] * gw[D_ + j];
    stops[(size_t)b * T_ + t] = 1.f / (1.f + expf(-acc));
  }
}

// ---------------- Host side ----------------
static inline size_t alignUp(size_t x, size_t a) { return (x + a - 1) & ~(a - 1); }

extern "C" void kernel_launch(void* const* d_in, const int* in_sizes, int n_in,
                              void* d_out, int out_size, void* d_ws, size_t ws_size,
                              hipStream_t stream) {
  (void)in_sizes; (void)n_in; (void)out_size;
  const float* memory   = (const float*)d_in[0];
  const float* mel_in   = (const float*)d_in[1];
  const float* pw1      = (const float*)d_in[2];
  const float* pw2      = (const float*)d_in[3];
  const float* aWx      = (const float*)d_in[4];
  const float* aWh      = (const float*)d_in[5];
  const float* aB       = (const float*)d_in[6];
  const float* wq       = (const float*)d_in[7];
  const float* wmem     = (const float*)d_in[8];
  const float* locconv  = (const float*)d_in[9];
  const float* wloc     = (const float*)d_in[10];
  const float* vatt     = (const float*)d_in[11];
  const float* dWx      = (const float*)d_in[12];
  const float* dWh      = (const float*)d_in[13];
  const float* dB       = (const float*)d_in[14];
  const float* projw    = (const float*)d_in[15];
  const float* projb    = (const float*)d_in[16];
  const float* gatew    = (const float*)d_in[17];
  const float* gateb    = (const float*)d_in[18];

  float* mels_out   = (float*)d_out;
  float* stops_out  = mels_out + (size_t)B_ * T_ * NMEL;
  float* aligns_out = stops_out + (size_t)B_ * T_;

  // --- workspace bump allocator ---
  char* ws = (char*)d_ws;
  size_t off = 0;
  auto alloc = [&](size_t bytes) {
    void* p = ws + off;
    off = alignUp(off + bytes, 256);
    return p;
  };
  typedef unsigned short us;
  us* w1t    = (us*)alloc((size_t)P_ * NMELP * 2);
  us* w2t    = (us*)alloc((size_t)P_ * P_ * 2);
  us* aWxt   = (us*)alloc((size_t)(4 * A_) * (P_ + E_) * 2);
  us* aWht   = (us*)alloc((size_t)(4 * A_) * A_ * 2);
  us* dWxt   = (us*)alloc((size_t)(4 * D_) * (A_ + E_) * 2);
  us* dWht   = (us*)alloc((size_t)(4 * D_) * D_ * 2);
  us* wqt    = (us*)alloc((size_t)AD_ * A_ * 2);
  us* wmemt  = (us*)alloc((size_t)AD_ * E_ * 2);
  us* membf  = (us*)alloc((size_t)B_ * S_ * E_ * 2);
  us* melbf  = (us*)alloc((size_t)B_ * NMELP * 2);
  us* p1bf   = (us*)alloc((size_t)B_ * P_ * 2);
  us* pbf    = (us*)alloc((size_t)B_ * P_ * 2);
  us* habf   = (us*)alloc((size_t)B_ * A_ * 2);
  us* hdbf   = (us*)alloc((size_t)B_ * D_ * 2);
  us* ctxbf  = (us*)alloc((size_t)B_ * E_ * 2);
  float* keys = (float*)alloc((size_t)B_ * S_ * AD_ * 4);
  float* z_a  = (float*)alloc((size_t)B_ * 4 * A_ * 4);
  float* z_d  = (float*)alloc((size_t)B_ * 4 * D_ * 4);
  float* h_a  = (float*)alloc((size_t)B_ * A_ * 4);
  float* c_a  = (float*)alloc((size_t)B_ * A_ * 4);
  float* h_d  = (float*)alloc((size_t)B_ * D_ * 4);
  float* c_d  = (float*)alloc((size_t)B_ * D_ * 4);
  float* ctx  = (float*)alloc((size_t)B_ * E_ * 4);
  float* qb   = (float*)alloc((size_t)B_ * AD_ * 4);
  float* fbuf = (float*)alloc((size_t)B_ * S_ * F_ * 4);
  float* ebuf = (float*)alloc((size_t)B_ * S_ * 4);
  float* wbuf = (float*)alloc((size_t)B_ * S_ * 4);
  float* wcum = (float*)alloc((size_t)B_ * S_ * 4);
  (void)ws_size;

  // --- zero the recurrent carries (graph-capture legal) ---
  hipMemsetAsync(h_a, 0, (size_t)B_ * A_ * 4, stream);
  hipMemsetAsync(c_a, 0, (size_t)B_ * A_ * 4, stream);
  hipMemsetAsync(h_d, 0, (size_t)B_ * D_ * 4, stream);
  hipMemsetAsync(c_d, 0, (size_t)B_ * D_ * 4, stream);
  hipMemsetAsync(ctx, 0, (size_t)B_ * E_ * 4, stream);
  hipMemsetAsync(wbuf, 0, (size_t)B_ * S_ * 4, stream);
  hipMemsetAsync(wcum, 0, (size_t)B_ * S_ * 4, stream);
  hipMemsetAsync(habf, 0, (size_t)B_ * A_ * 2, stream);
  hipMemsetAsync(hdbf, 0, (size_t)B_ * D_ * 2, stream);
  hipMemsetAsync(ctxbf, 0, (size_t)B_ * E_ * 2, stream);

  // --- once-per-launch weight conversion (fp32 -> transposed bf16) ---
  auto launchT = [&](const float* W, us* Wt, int K, int N, int Kpad) {
    int n = N * Kpad;
    k_transpose_bf<<<(n + 255) / 256, 256, 0, stream>>>(W, Wt, K, N, Kpad);
  };
  launchT(pw1, w1t, NMEL, P_, NMELP);
  launchT(pw2, w2t, P_, P_, P_);
  launchT(aWx, aWxt, P_ + E_, 4 * A_, P_ + E_);
  launchT(aWh, aWht, A_, 4 * A_, A_);
  launchT(dWx, dWxt, A_ + E_, 4 * D_, A_ + E_);
  launchT(dWh, dWht, D_, 4 * D_, D_);
  launchT(wq, wqt, A_, AD_, A_);
  launchT(wmem, wmemt, E_, AD_, E_);
  {
    int n = B_ * S_ * E_;
    k_convert_bf<<<(n + 255) / 256, 256, 0, stream>>>(memory, membf, n);
  }

  auto gemm = [&](const us* A0, int lda0, const us* W0, int ldw0, int K0,
                  const us* A1, int lda1, const us* W1, int ldw1, int K1,
                  const us* A2, int lda2, const us* W2, int ldw2, int K2,
                  const float* bias, float* outF, us* outH, int M, int N, int relu) {
    GemmArgs ga;
    ga.A[0] = A0; ga.W[0] = W0; ga.lda[0] = lda0; ga.ldw[0] = ldw0; ga.K[0] = K0;
    ga.A[1] = A1; ga.W[1] = W1; ga.lda[1] = lda1; ga.ldw[1] = ldw1; ga.K[1] = K1;
    ga.A[2] = A2; ga.W[2] = W2; ga.lda[2] = lda2; ga.ldw[2] = ldw2; ga.K[2] = K2;
    ga.bias = bias; ga.outF = outF; ga.outH = outH; ga.N = N; ga.relu = relu;
    dim3 grid(N / 64, M / 64);
    k_wmma_gemm<<<grid, 256, 0, stream>>>(ga);
  };

  // keys = memory @ w_memory   [B*S, E] x [E, AD]
  gemm(membf, E_, wmemt, E_, E_, nullptr, 0, nullptr, 0, 0, nullptr, 0, nullptr, 0, 0,
       nullptr, keys, nullptr, B_ * S_, AD_, 0);

  // --- sequential decode ---
  for (int t = 0; t < T_; ++t) {
    // prenet
    k_mel_slice<<<(B_ * NMELP + 255) / 256, 256, 0, stream>>>(mel_in, melbf, t);
    gemm(melbf, NMELP, w1t, NMELP, NMELP, nullptr, 0, nullptr, 0, 0, nullptr, 0, nullptr, 0, 0,
         nullptr, nullptr, p1bf, B_, P_, 1);
    gemm(p1bf, P_, w2t, P_, P_, nullptr, 0, nullptr, 0, 0, nullptr, 0, nullptr, 0, 0,
         nullptr, nullptr, pbf, B_, P_, 1);
    // attention LSTM: z = [p, ctx] @ Wx + h_a @ Wh + b
    gemm(pbf, P_, aWxt, P_ + E_, P_,
         ctxbf, E_, aWxt + P_, P_ + E_, E_,
         habf, A_, aWht, A_, A_,
         aB, z_a, nullptr, B_, 4 * A_, 0);
    k_lstm_pw<<<(B_ * A_ + 255) / 256, 256, 0, stream>>>(z_a, c_a, h_a, habf, A_);
    // query projection
    gemm(habf, A_, wqt, A_, A_, nullptr, 0, nullptr, 0, 0, nullptr, 0, nullptr, 0, 0,
         nullptr, qb, nullptr, B_, AD_, 0);
    // location-sensitive attention
    k_locconv<<<(B_ * S_ * F_ + 255) / 256, 256, 0, stream>>>(wbuf, wcum, locconv, fbuf);
    k_energy<<<B_ * S_, AD_, 0, stream>>>(qb, keys, fbuf, wloc, vatt, ebuf);
    k_softmax_ctx<<<B_, S_, 0, stream>>>(ebuf, wbuf, wcum, aligns_out, memory, ctx, ctxbf, t);
    // decoder LSTM: z = [h_a, ctx] @ Wx + h_d @ Wh + b
    gemm(habf, A_, dWxt, A_ + E_, A_,
         ctxbf, E_, dWxt + A_, A_ + E_, E_,
         hdbf, D_, dWht, D_, D_,
         dB, z_d, nullptr, B_, 4 * D_, 0);
    k_lstm_pw<<<(B_ * D_ + 255) / 256, 256, 0, stream>>>(z_d, c_d, h_d, hdbf, D_);
    // projection + stop gate
    k_proj<<<B_, 96, 0, stream>>>(h_d, ctx, projw, projb, gatew, gateb,
                                  mels_out, stops_out, t);
  }
}
